// DTD_8083128451212
// MI455X (gfx1250) — compile-verified, matrix operand order checked by reading, not run on hardware
//
#include <hip/hip_runtime.h>
#include <hip/hip_bf16.h>
#include <stdint.h>

// ---------------------------------------------------------------------------
// Types for CDNA5 WMMA (wave32): bf16 A/B fragments, f32 accumulator.
// ---------------------------------------------------------------------------
typedef __attribute__((ext_vector_type(16))) __bf16 v16bf;
typedef __attribute__((ext_vector_type(8)))  __bf16 v8bf;   // 16 bytes
typedef __attribute__((ext_vector_type(8)))  float  v8f;

__device__ __forceinline__ __bf16 f2bf(float x) {
    union { float f; uint32_t u; } v; v.f = x;
    uint32_t r = v.u + 0x7FFFu + ((v.u >> 16) & 1u);   // round-to-nearest-even
    uint16_t h = (uint16_t)(r >> 16);
    __bf16 out;
    __builtin_memcpy(&out, &h, 2);
    return out;
}

__device__ __forceinline__ float sigmoidf_(float x) {
    return 1.0f / (1.0f + __expf(-x));
}

// Unconditional 16-bit fragment load: two b128 loads (base must be in-bounds;
// caller guarantees padding). Byte offset is 16B-aligned by construction.
__device__ __forceinline__ v16bf load_frag(const __bf16* __restrict__ base,
                                           int ld, int row, int k0, int kb) {
    const v8bf* p = (const v8bf*)(base + (long long)row * ld + k0 + kb);
    v8bf lo = p[0];
    v8bf hi = p[2];      // +16 elements
    return __builtin_shufflevector(lo, hi, 0, 1, 2, 3, 4, 5, 6, 7,
                                           8, 9, 10, 11, 12, 13, 14, 15);
}

// ---------------------------------------------------------------------------
// Generic batched GEMM:  O[m,n] = sum_k A[m,k] * W[n,k]  (i.e. A @ W^T)
// A, W bf16 row-major and PADDED so every tile read is in-bounds:
//   - A must have >= tilesM*16 readable rows per batch
//   - W must have >= tilesN4*64 readable rows per batch
//   - K multiple of 32; lda/ldw multiples of 8 elements
// Cinit (f32) requires exact M,N tiling (used only for LSTM steps: 128x1024).
// One wave computes a 16x64 strip (4 WMMA tiles sharing one A fragment).
// ---------------------------------------------------------------------------
__global__ void gemm_bf16_wmma(const __bf16* __restrict__ Abase, int lda, long long strideA,
                               const __bf16* __restrict__ Wbase, int ldw, long long strideW,
                               const float*  __restrict__ Cinit, int ldci,
                               const float*  __restrict__ bias,
                               float*   __restrict__ Obase,
                               __bf16*  __restrict__ Obf,
                               int ldo, long long strideO,
                               int M, int N, int K,
                               int tilesM, int tilesN4, int nBatch)
{
    const int lane   = threadIdx.x & 31;
    const int waveId = (int)((blockIdx.x * (long long)blockDim.x + threadIdx.x) >> 5);
    const int perBatch = tilesM * tilesN4;
    const int batch = waveId / perBatch;
    if (batch >= nBatch) return;                    // wave-uniform exit
    const int tile = waveId - batch * perBatch;
    const int tm   = tile / tilesN4;
    const int tn4  = tile - tm * tilesN4;
    const int row0 = tm * 16;
    const int col0 = tn4 * 64;

    const __bf16* A = Abase + (long long)batch * strideA;
    const __bf16* W = Wbase + (long long)batch * strideW;

    // accumulator C/D layout: element e -> row = row0 + e + 8*(lane>>4), col = col0 + (lane&15)
    const int crow = (lane >> 4) * 8;
    const int ccol = lane & 15;

    v8f acc[4];
    #pragma unroll
    for (int nt = 0; nt < 4; ++nt) {
        v8f a = {};
        const int c0 = col0 + nt * 16 + ccol;
        if (Cinit) {
            const float* cp = Cinit + (long long)(row0 + crow) * ldci + c0;
            #pragma unroll
            for (int e = 0; e < 8; ++e) a[e] = cp[(long long)e * ldci];
        } else if (bias) {
            const float bv = (c0 < N) ? bias[c0] : 0.0f;
            #pragma unroll
            for (int e = 0; e < 8; ++e) a[e] = bv;
        }
        acc[nt] = a;
    }

    // A/B fragment layout (16-bit, 16x32): lane<16 holds K {0..7,16..23} of row lane,
    // lane>=16 holds K {8..15,24..31} of row lane-16.
    const int ar = lane & 15;
    const int kb = (lane >> 4) * 8;

    for (int k0 = 0; k0 < K; k0 += 32) {
        const v16bf af = load_frag(A, lda, row0 + ar, k0, kb);
        __builtin_prefetch((const void*)(A + (long long)(row0 + ar) * lda + k0 + kb + 32), 0, 1);
        #pragma unroll
        for (int nt = 0; nt < 4; ++nt) {
            const v16bf bfrag = load_frag(W, ldw, col0 + nt * 16 + ar, k0, kb);
            acc[nt] = __builtin_amdgcn_wmma_f32_16x16x32_bf16(
                false, af, false, bfrag, (short)0, acc[nt], false, false);
        }
    }

    float*  O  = Obase ? Obase + (long long)batch * strideO : nullptr;
    __bf16* Ob = Obf   ? Obf   + (long long)batch * strideO : nullptr;
    #pragma unroll
    for (int nt = 0; nt < 4; ++nt) {
        const int c0 = col0 + nt * 16 + ccol;
        if (c0 >= N) continue;
        #pragma unroll
        for (int e = 0; e < 8; ++e) {
            const int r = row0 + crow + e;
            if (r >= M) continue;
            const long long off = (long long)r * ldo + c0;
            const float v = acc[nt][e];
            if (O)  O[off]  = v;
            if (Ob) Ob[off] = f2bf(v);
        }
    }
}

// ---------------------------------------------------------------------------
// Elementwise helpers
// ---------------------------------------------------------------------------
__global__ void f32_to_bf16_kernel(const float* __restrict__ in, __bf16* __restrict__ out, long long n) {
    long long i = (long long)blockIdx.x * blockDim.x + threadIdx.x;
    if (i < n) out[i] = f2bf(in[i]);
}

__global__ void zero_u32_kernel(uint32_t* __restrict__ p, long long n) {
    long long i = (long long)blockIdx.x * blockDim.x + threadIdx.x;
    if (i < n) p[i] = 0u;
}

__global__ void add_vec_kernel(const float* __restrict__ a, const float* __restrict__ b,
                               float* __restrict__ o, int n) {
    int i = blockIdx.x * blockDim.x + threadIdx.x;
    if (i < n) o[i] = a[i] + b[i];
}

// ---------------------------------------------------------------------------
// Attention normalization: per (b,t), A_norm = A / sum_hw(A). 256 spatial elems.
// Writes f32 (b*25+t rows, for gather) and bf16 into PADDED layout (b*32+t rows).
// ---------------------------------------------------------------------------
__global__ void normalize_attn_kernel(const float* __restrict__ A,
                                      float* __restrict__ A_norm,
                                      __bf16* __restrict__ A_bf)
{
    const int bt  = blockIdx.x;          // b*25 + t
    const int b   = bt / 25;
    const int t   = bt - b * 25;
    const int tid = threadIdx.x;         // 0..255 = hw
    const float v = A[(long long)bt * 256 + tid];
    __shared__ float red[256];
    red[tid] = v;
    __syncthreads();
    for (int s = 128; s > 0; s >>= 1) {
        if (tid < s) red[tid] += red[tid + s];
        __syncthreads();
    }
    const float r = v / red[0];
    A_norm[(long long)bt * 256 + tid] = r;
    A_bf[((long long)b * 32 + t) * 256 + tid] = f2bf(r);
}

// ---------------------------------------------------------------------------
// LSTM pointwise: gates G (128 x 1024, order i,f,g,o), fp32 c state,
// writes h (f32, into hf[t]) and bf16 h state for the next step GEMM.
// ---------------------------------------------------------------------------
__global__ void lstm_pointwise_kernel(const float* __restrict__ G,
                                      float* __restrict__ c_state,
                                      float* __restrict__ h_out_f32,
                                      __bf16* __restrict__ h_bf)
{
    const int idx = blockIdx.x * blockDim.x + threadIdx.x;   // 128*256
    if (idx >= 128 * 256) return;
    const int b = idx >> 8;
    const int j = idx & 255;
    const float* g = G + (long long)b * 1024;
    const float gi = sigmoidf_(g[j]);
    const float gf = sigmoidf_(g[256 + j]);
    const float gg = tanhf(g[512 + j]);
    const float go = sigmoidf_(g[768 + j]);
    const float c  = gf * c_state[idx] + gi * gg;
    c_state[idx] = c;
    const float h = go * tanhf(c);
    h_out_f32[idx] = h;
    h_bf[idx] = f2bf(h);
}

// ---------------------------------------------------------------------------
// Build GRU inputs xs = [hf | hb | emb(prev_token)] in bf16. (T*B, 1024)
// ---------------------------------------------------------------------------
__global__ void build_xs_kernel(const float* __restrict__ hf,
                                const float* __restrict__ hb,
                                const float* __restrict__ char_emb,
                                const int*   __restrict__ text,
                                __bf16* __restrict__ xs)
{
    const long long idx = (long long)blockIdx.x * blockDim.x + threadIdx.x;
    if (idx >= (long long)3200 * 1024) return;
    const int col = (int)(idx & 1023);
    const int row = (int)(idx >> 10);        // t*128 + b
    const int t = row >> 7;
    const int b = row & 127;
    float v;
    if (col < 256) {
        v = hf[(long long)row * 256 + col];
    } else if (col < 512) {
        v = hb[(long long)row * 256 + (col - 256)];
    } else {
        const int prev = (t == 0) ? 0 : text[b * 25 + (t - 1)];
        v = char_emb[(long long)prev * 512 + (col - 512)];
    }
    xs[idx] = f2bf(v);
}

// ---------------------------------------------------------------------------
// GRU pointwise: gi precomputed (row t*128+b, ld 1536), Gh = h@Whh^T + b_hh.
// r,z,n gate order. Writes f32 h state, bf16 h state, and bf16 gru_res row.
// ---------------------------------------------------------------------------
__global__ void gru_pointwise_kernel(const float* __restrict__ Gh,
                                     const float* __restrict__ gi_t,
                                     float*  __restrict__ h_state,
                                     __bf16* __restrict__ h_bf,
                                     __bf16* __restrict__ gru_bf_t)
{
    const int idx = blockIdx.x * blockDim.x + threadIdx.x;   // 128*512
    if (idx >= 128 * 512) return;
    const int b = idx >> 9;
    const int j = idx & 511;
    const float* gh = Gh   + (long long)b * 1536;
    const float* gi = gi_t + (long long)b * 1536;
    const float r = sigmoidf_(gi[j]        + gh[j]);
    const float z = sigmoidf_(gi[512 + j]  + gh[512 + j]);
    const float n = tanhf(gi[1024 + j] + r * gh[1024 + j]);
    const float h = (1.0f - z) * n + z * h_state[idx];
    h_state[idx] = h;
    const __bf16 hbv = f2bf(h);
    h_bf[idx] = hbv;
    gru_bf_t[(long long)b * 512 + j] = hbv;
}

// ---------------------------------------------------------------------------
// Prefix scan of text_length (128 entries) -> offs[0..128]
// ---------------------------------------------------------------------------
__global__ void scan_lengths_kernel(const int* __restrict__ len, int* __restrict__ offs) {
    if (threadIdx.x == 0 && blockIdx.x == 0) {
        int acc = 0;
        for (int b = 0; b < 128; ++b) { offs[b] = acc; acc += len[b]; }
        offs[128] = acc;
    }
}

// ---------------------------------------------------------------------------
// Ragged gather: out = [out_res (lenText,38) | out_attns (lenText,256)] flat.
// ---------------------------------------------------------------------------
__global__ void gather_out_kernel(const float* __restrict__ logits,   // (T*B, 38)
                                  const float* __restrict__ A_norm,   // (B*T, 256)
                                  const int* __restrict__ len,
                                  const int* __restrict__ offs,
                                  float* __restrict__ out)
{
    const int bt = blockIdx.x;           // b*25 + t
    const int b = bt / 25;
    const int t = bt - b * 25;
    if (t >= len[b]) return;
    const int row   = offs[b] + t;
    const int total = offs[128];
    const int k = threadIdx.x;           // 0..255
    if (k < 38)
        out[(long long)row * 38 + k] = logits[((long long)t * 128 + b) * 38 + k];
    out[(long long)total * 38 + (long long)row * 256 + k] =
        A_norm[((long long)b * 25 + t) * 256 + k];
}

// ---------------------------------------------------------------------------
// Host side
// ---------------------------------------------------------------------------
static void launch_gemm(hipStream_t s,
                        const __bf16* A, int lda, long long sA,
                        const __bf16* W, int ldw, long long sW,
                        const float* Ci, int ldci, const float* bias,
                        float* O, __bf16* Ob, int ldo, long long sO,
                        int M, int N, int K, int nBatch)
{
    const int tilesM  = (M + 15) / 16;
    const int tilesN4 = (N + 63) / 64;
    const long long waves = (long long)nBatch * tilesM * tilesN4;
    const int blocks = (int)((waves + 7) / 8);
    gemm_bf16_wmma<<<blocks, 256, 0, s>>>(A, lda, sA, W, ldw, sW, Ci, ldci, bias,
                                          O, Ob, ldo, sO, M, N, K,
                                          tilesM, tilesN4, nBatch);
}

static void launch_cvt(hipStream_t s, const float* in, __bf16* out, long long n) {
    f32_to_bf16_kernel<<<(int)((n + 255) / 256), 256, 0, s>>>(in, out, n);
}

static void launch_zero(hipStream_t s, void* p, long long bytes) {
    long long n = bytes / 4;
    zero_u32_kernel<<<(int)((n + 255) / 256), 256, 0, s>>>((uint32_t*)p, n);
}

extern "C" void kernel_launch(void* const* d_in, const int* in_sizes, int n_in,
                              void* d_out, int out_size, void* d_ws, size_t ws_size,
                              hipStream_t stream)
{
    (void)in_sizes; (void)n_in; (void)out_size; (void)ws_size;
    const float* feature  = (const float*)d_in[0];   // (128,512,8,32)
    const float* A        = (const float*)d_in[1];   // (128,25,8,32)
    const int*   text     = (const int*)  d_in[2];   // (128,25)
    const int*   text_len = (const int*)  d_in[3];   // (128,)
    const float* W_ih_f   = (const float*)d_in[4];
    const float* W_hh_f   = (const float*)d_in[5];
    const float* b_ih_f   = (const float*)d_in[6];
    const float* b_hh_f   = (const float*)d_in[7];
    const float* W_ih_b   = (const float*)d_in[8];
    const float* W_hh_b   = (const float*)d_in[9];
    const float* b_ih_b   = (const float*)d_in[10];
    const float* b_hh_b   = (const float*)d_in[11];
    const float* W_ih_g   = (const float*)d_in[12];
    const float* W_hh_g   = (const float*)d_in[13];
    const float* b_ih_g   = (const float*)d_in[14];
    const float* b_hh_g   = (const float*)d_in[15];
    const float* W_gen    = (const float*)d_in[16];
    const float* b_gen    = (const float*)d_in[17];
    const float* char_emb = (const float*)d_in[18];
    float* out = (float*)d_out;

    // ---- bump allocator over d_ws ----
    char* ws = (char*)d_ws;
    size_t off = 0;
    auto alloc = [&](size_t bytes) -> void* {
        void* p = ws + off;
        off += (bytes + 255) & ~(size_t)255;
        return p;
    };

    // F_bf (used only through pooling) aliased with gi (used only after xs build)
    char*   unionBase = (char*)alloc(128ll * 512 * 256 * 2);   // 33.5 MB >= gi 19.7 MB
    __bf16* F_bf  = (__bf16*)unionBase;
    float*  gi    = (float*) unionBase;

    float*  A_norm  = (float*) alloc(128ll * 25 * 256 * 4);
    __bf16* A_bf    = (__bf16*)alloc(128ll * 32 * 256 * 2);    // padded to 32 rows/batch
    __bf16* C_bf    = (__bf16*)alloc(3200ll * 512 * 2);
    __bf16* Wihf_bf = (__bf16*)alloc(1024ll * 512 * 2);
    __bf16* Whhf_bf = (__bf16*)alloc(1024ll * 256 * 2);
    __bf16* Wihb_bf = (__bf16*)alloc(1024ll * 512 * 2);
    __bf16* Whhb_bf = (__bf16*)alloc(1024ll * 256 * 2);
    __bf16* Wihg_bf = (__bf16*)alloc(1536ll * 1024 * 2);
    __bf16* Whhg_bf = (__bf16*)alloc(1536ll * 512 * 2);
    __bf16* Wgen_bf = (__bf16*)alloc(64ll * 512 * 2);          // padded to 64 rows
    float*  bias_f  = (float*) alloc(1024 * 4);
    float*  bias_b  = (float*) alloc(1024 * 4);
    float*  Xf      = (float*) alloc(3200ll * 1024 * 4);
    float*  Xb      = (float*) alloc(3200ll * 1024 * 4);
    float*  hf      = (float*) alloc(3200ll * 256 * 4);
    float*  hb      = (float*) alloc(3200ll * 256 * 4);
    __bf16* hbf_f   = (__bf16*)alloc(128ll * 256 * 2);
    __bf16* hbf_b   = (__bf16*)alloc(128ll * 256 * 2);
    float*  c_f     = (float*) alloc(128ll * 256 * 4);
    float*  c_b     = (float*) alloc(128ll * 256 * 4);
    float*  Gf      = (float*) alloc(128ll * 1024 * 4);
    float*  Gb      = (float*) alloc(128ll * 1024 * 4);
    __bf16* xs_bf   = (__bf16*)alloc(3200ll * 1024 * 2);
    float*  Gh      = (float*) alloc(128ll * 1536 * 4);
    float*  gruh    = (float*) alloc(128ll * 512 * 4);
    __bf16* gruh_bf = (__bf16*)alloc(128ll * 512 * 2);
    __bf16* gru_bf  = (__bf16*)alloc(3200ll * 512 * 2);
    float*  logits  = (float*) alloc(3200ll * 38 * 4);
    int*    offs    = (int*)   alloc(129 * 4);

    // ---- 0. zero padded operand buffers (pad rows must read as 0) ----
    launch_zero(stream, A_bf,    128ll * 32 * 256 * 2);
    launch_zero(stream, Wgen_bf, 64ll * 512 * 2);

    // ---- 1. convert operands to bf16 ----
    launch_cvt(stream, feature, F_bf, 128ll * 512 * 256);
    launch_cvt(stream, W_ih_f, Wihf_bf, 1024ll * 512);
    launch_cvt(stream, W_hh_f, Whhf_bf, 1024ll * 256);
    launch_cvt(stream, W_ih_b, Wihb_bf, 1024ll * 512);
    launch_cvt(stream, W_hh_b, Whhb_bf, 1024ll * 256);
    launch_cvt(stream, W_ih_g, Wihg_bf, 1536ll * 1024);
    launch_cvt(stream, W_hh_g, Whhg_bf, 1536ll * 512);
    launch_cvt(stream, W_gen,  Wgen_bf, 38ll * 512);
    add_vec_kernel<<<4, 256, 0, stream>>>(b_ih_f, b_hh_f, bias_f, 1024);
    add_vec_kernel<<<4, 256, 0, stream>>>(b_ih_b, b_hh_b, bias_b, 1024);

    // ---- 2. normalize attention ----
    normalize_attn_kernel<<<3200, 256, 0, stream>>>(A, A_norm, A_bf);

    // ---- 3. pooled features C[t,b,c] via batched WMMA GEMM (per b: 25x512x256) ----
    // out element (t,c) for batch b at t*65536 + b*512 + c  -> ldo=65536, strideO=512
    launch_gemm(stream, A_bf, 256, 32ll * 256,
                F_bf, 256, 512ll * 256,
                nullptr, 0, nullptr,
                nullptr, C_bf, 128 * 512, 512,
                25, 512, 256, 128);

    // ---- 4. LSTM input projections: X = C @ W_ih^T + (b_ih + b_hh) ----
    launch_gemm(stream, C_bf, 512, 0, Wihf_bf, 512, 0, nullptr, 0, bias_f,
                Xf, nullptr, 1024, 0, 3200, 1024, 512, 1);
    launch_gemm(stream, C_bf, 512, 0, Wihb_bf, 512, 0, nullptr, 0, bias_b,
                Xb, nullptr, 1024, 0, 3200, 1024, 512, 1);

    // ---- 5. zero recurrent state ----
    launch_zero(stream, c_f,     128ll * 256 * 4);
    launch_zero(stream, c_b,     128ll * 256 * 4);
    launch_zero(stream, hbf_f,   128ll * 256 * 2);
    launch_zero(stream, hbf_b,   128ll * 256 * 2);
    launch_zero(stream, gruh,    128ll * 512 * 4);
    launch_zero(stream, gruh_bf, 128ll * 512 * 2);

    // ---- 6. bidirectional LSTM recurrence (25 steps each dir) ----
    for (int s = 0; s < 25; ++s) {
        const int tf = s;          // forward time
        const int tb = 24 - s;     // backward time
        launch_gemm(stream, hbf_f, 256, 0, Whhf_bf, 256, 0,
                    Xf + (long long)tf * 128 * 1024, 1024, nullptr,
                    Gf, nullptr, 1024, 0, 128, 1024, 256, 1);
        lstm_pointwise_kernel<<<128, 256, 0, stream>>>(
            Gf, c_f, hf + (long long)tf * 128 * 256, hbf_f);
        launch_gemm(stream, hbf_b, 256, 0, Whhb_bf, 256, 0,
                    Xb + (long long)tb * 128 * 1024, 1024, nullptr,
                    Gb, nullptr, 1024, 0, 128, 1024, 256, 1);
        lstm_pointwise_kernel<<<128, 256, 0, stream>>>(
            Gb, c_b, hb + (long long)tb * 128 * 256, hbf_b);
    }

    // ---- 7. GRU inputs: xs = [hf|hb|emb(prev_tok)] and gi = xs @ W_ih_g^T + b_ih_g ----
    build_xs_kernel<<<12800, 256, 0, stream>>>(hf, hb, char_emb, text, xs_bf);
    launch_gemm(stream, xs_bf, 1024, 0, Wihg_bf, 1024, 0, nullptr, 0, b_ih_g,
                gi, nullptr, 1536, 0, 3200, 1536, 1024, 1);

    // ---- 8. GRU recurrence ----
    for (int t = 0; t < 25; ++t) {
        launch_gemm(stream, gruh_bf, 512, 0, Whhg_bf, 512, 0, nullptr, 0, b_hh_g,
                    Gh, nullptr, 1536, 0, 128, 1536, 512, 1);
        gru_pointwise_kernel<<<256, 256, 0, stream>>>(
            Gh, gi + (long long)t * 128 * 1536, gruh, gruh_bf,
            gru_bf + (long long)t * 128 * 512);
    }

    // ---- 9. generator: logits = gru_res @ W_gen^T + b_gen ----
    launch_gemm(stream, gru_bf, 512, 0, Wgen_bf, 512, 0, nullptr, 0, b_gen,
                logits, nullptr, 38, 0, 3200, 38, 512, 1);

    // ---- 10. ragged assembly ----
    scan_lengths_kernel<<<1, 32, 0, stream>>>(text_len, offs);
    gather_out_kernel<<<3200, 256, 0, stream>>>(logits, A_norm, text_len, offs, out);
}
